// SparseMoE_89721866813908
// MI455X (gfx1250) — matrix operand order repaired and checked by reading, not compiled, not run on hardware
//
#include <hip/hip_runtime.h>
#include <hip/hip_bf16.h>

typedef __bf16 bf16;
typedef __attribute__((ext_vector_type(8)))  float        v8f;
typedef __attribute__((ext_vector_type(8)))  bf16         v8bf;
typedef __attribute__((ext_vector_type(16))) bf16         v16bf;
typedef __attribute__((ext_vector_type(4)))  unsigned int v4u;
typedef __attribute__((ext_vector_type(8)))  unsigned int v8u;

#define DIMC 1024
#define HID  1536
#define NEXP 8
#define NTOK 8192
#define TOPK 2

// ---------------------------------------------------------------------------
// WMMA helper: D = A(16x32 bf16) x B(32x16 bf16) + C(16x16 f32)
// ---------------------------------------------------------------------------
static __device__ inline v8f wmma_bf16(v16bf a, v16bf b, v8f c) {
  return __builtin_amdgcn_wmma_f32_16x16x32_bf16(
      /*neg_a=*/false, a, /*neg_b=*/false, b,
      /*c_mod=*/(short)0, c, /*reuse_a=*/false, /*reuse_b=*/false);
}

// A-tile load: lane half holds K [base, base+16) contiguous (two 16B chunks).
static __device__ inline v16bf load_a16(const bf16* p) {
  v8bf lo = *(const v8bf*)(p);
  v8bf hi = *(const v8bf*)(p + 8);
  return __builtin_shufflevector(lo, hi, 0, 1, 2, 3, 4, 5, 6, 7,
                                 8, 9, 10, 11, 12, 13, 14, 15);
}

// ---------------------------------------------------------------------------
// fp32 -> bf16 straight conversion (x)
// ---------------------------------------------------------------------------
__global__ void f32_to_bf16_kernel(const float* __restrict__ in,
                                   bf16* __restrict__ out, size_t n) {
  size_t i = (size_t)blockIdx.x * blockDim.x + threadIdx.x;
  size_t stride = (size_t)gridDim.x * blockDim.x;
  for (; i < n; i += stride) out[i] = (bf16)in[i];
}

// ---------------------------------------------------------------------------
// Tiled transpose + convert: src [rows][cols] f32 -> dst [cols][rows] bf16.
// One z-slice per expert. Coalesced on both sides via 32x32 LDS tile.
// ---------------------------------------------------------------------------
__global__ __launch_bounds__(256) void transpose_bf16_kernel(
    const float* __restrict__ src, bf16* __restrict__ dst, int rows, int cols) {
  __shared__ float tile[32][33];
  size_t slice = (size_t)rows * cols;
  const float* s = src + (size_t)blockIdx.z * slice;
  bf16* d = dst + (size_t)blockIdx.z * slice;
  int c0 = blockIdx.x * 32;
  int r0 = blockIdx.y * 32;
  int tx = threadIdx.x;
  for (int rr = threadIdx.y; rr < 32; rr += 8)
    tile[rr][tx] = s[(size_t)(r0 + rr) * cols + c0 + tx];
  __syncthreads();
  for (int cc = threadIdx.y; cc < 32; cc += 8)
    d[(size_t)(c0 + cc) * rows + r0 + tx] = (bf16)tile[tx][cc];
}

// ---------------------------------------------------------------------------
// Router: one wave32 per token. Logits = x[n] . Wr[:,e], top-2 + softmax.
// ---------------------------------------------------------------------------
__global__ __launch_bounds__(256) void router_kernel(
    const float* __restrict__ x, const float* __restrict__ Wr,
    int* __restrict__ tk_idx, float* __restrict__ tk_w) {
  int wave = threadIdx.x >> 5;
  int lane = threadIdx.x & 31;
  int n = blockIdx.x * 8 + wave;
  if (n >= NTOK) return;
  const float* xr = x + (size_t)n * DIMC;
  float lg[NEXP];
#pragma unroll
  for (int e = 0; e < NEXP; ++e) lg[e] = 0.f;
  for (int c = lane; c < DIMC; c += 32) {
    float xv = xr[c];
    const float* wr = Wr + (size_t)c * NEXP;
#pragma unroll
    for (int e = 0; e < NEXP; ++e) lg[e] += xv * wr[e];
  }
#pragma unroll
  for (int e = 0; e < NEXP; ++e)
    for (int off = 16; off > 0; off >>= 1)
      lg[e] += __shfl_xor(lg[e], off, 32);
  if (lane == 0) {
    int i0 = 0;
    for (int e = 1; e < NEXP; ++e)
      if (lg[e] > lg[i0]) i0 = e;
    int i1 = (i0 == 0) ? 1 : 0;
    for (int e = 0; e < NEXP; ++e)
      if (e != i0 && lg[e] > lg[i1]) i1 = e;
    float e0 = __expf(lg[i0] - lg[i0]);
    float e1 = __expf(lg[i1] - lg[i0]);
    float inv = 1.f / (e0 + e1);
    tk_idx[n * 2 + 0] = i0;
    tk_idx[n * 2 + 1] = i1;
    tk_w[n * 2 + 0] = e0 * inv;
    tk_w[n * 2 + 1] = e1 * inv;
  }
}

// ---------------------------------------------------------------------------
// Scatter tokens into per-expert lists.
// ---------------------------------------------------------------------------
__global__ void scatter_kernel(const int* __restrict__ tk_idx,
                               const float* __restrict__ tk_w,
                               int* __restrict__ cnt, int* __restrict__ list,
                               float* __restrict__ lw) {
  int i = blockIdx.x * blockDim.x + threadIdx.x;
  if (i >= NTOK * TOPK) return;
  int e = tk_idx[i];
  int pos = atomicAdd(&cnt[e], 1);
  list[(size_t)e * NTOK + pos] = i >> 1;
  lw[(size_t)e * NTOK + pos] = tk_w[i];
}

// ---------------------------------------------------------------------------
// Fused expert SwiGLU GEMM. Block = 8 waves, one (16-token tile, expert).
//   TDM gather: 16 scattered token rows of x (bf16) -> LDS x-tile (32 KB)
//   Phase 1: act(16x1536) = silu(x W1) * (x W3)   [WMMA bf16, A from LDS]
//   Phase 2: y(16x1024)   = act W2                [WMMA bf16, A from LDS]
//   out[tok] += comb_w * y   (global f32 atomics; 2 experts per token)
// Weights are pre-transposed so each B-tile lane load is 32B contiguous.
// ---------------------------------------------------------------------------
__global__ __launch_bounds__(256) void moe_gemm_kernel(
    const bf16* __restrict__ xb, const bf16* __restrict__ w1t,
    const bf16* __restrict__ w3t, const bf16* __restrict__ w2t,
    const int* __restrict__ cnt, const int* __restrict__ list,
    const float* __restrict__ lw, float* __restrict__ out) {
  __shared__ bf16 xtile[16 * DIMC];  // 32 KB, gathered token rows
  __shared__ bf16 act[16 * HID];     // 48 KB
  __shared__ int toks[16];
  __shared__ float tws[16];

  int e = blockIdx.y;
  int tile = blockIdx.x;
  int nt = cnt[e];
  if (tile * 16 >= nt) return;  // block-uniform exit

  int tid = threadIdx.x;
  if (tid < 16) {
    int i = tile * 16 + tid;
    bool valid = i < nt;
    toks[tid] = valid ? list[(size_t)e * NTOK + i] : list[(size_t)e * NTOK];
    tws[tid] = valid ? lw[(size_t)e * NTOK + i] : 0.f;
  }

  // ---- TDM gather: 16 token rows (1024 bf16 each) -> LDS, wave 0 issues ----
  if (tid < 32) {
    unsigned idx16[16];
#pragma unroll
    for (int i = 0; i < 16; ++i) {
      int ii = tile * 16 + i;
      if (ii >= nt) ii = nt - 1;
      idx16[i] = (unsigned)list[(size_t)e * NTOK + ii];
    }
    unsigned long long gaddr = (unsigned long long)(uintptr_t)xb;
    unsigned ldsaddr = (unsigned)(uintptr_t)(&xtile[0]);

    v4u g0;
    g0[0] = 0x80000001u;  // count=1, gather_mode=1, 16-bit indices
    g0[1] = ldsaddr;      // lds_addr
    g0[2] = (unsigned)(gaddr & 0xFFFFFFFFu);                     // gaddr lo
    g0[3] = (unsigned)((gaddr >> 32) & 0x01FFFFFFu) | 0x80000000u;  // type=2

    v8u g1;
    g1[0] = 0x00010000u;                     // wg_mask=0, data_size=1 (2B)
    g1[1] = ((unsigned)DIMC & 0xFFFFu) << 16;   // tensor_dim0 -> bits 63:48
    g1[2] = (((unsigned)DIMC >> 16) & 0xFFFFu) |
            (((unsigned)NTOK & 0xFFFFu) << 16);  // dim0 hi | tensor_dim1 lo
    g1[3] = (((unsigned)NTOK >> 16) & 0xFFFFu) |
            (((unsigned)DIMC & 0xFFFFu) << 16);  // dim1 hi | tile_dim0=1024
    g1[4] = 16u;                             // tile_dim1 = #gather indices
    g1[5] = (unsigned)DIMC;                  // tensor_dim0_stride lo 32
    g1[6] = 0u;                              // stride hi | dim1_stride lo
    g1[7] = 0u;

    v4u g2, g3;
    g2[0] = idx16[0] | (idx16[1] << 16);
    g2[1] = idx16[2] | (idx16[3] << 16);
    g2[2] = idx16[4] | (idx16[5] << 16);
    g2[3] = idx16[6] | (idx16[7] << 16);
    g3[0] = idx16[8] | (idx16[9] << 16);
    g3[1] = idx16[10] | (idx16[11] << 16);
    g3[2] = idx16[12] | (idx16[13] << 16);
    g3[3] = idx16[14] | (idx16[15] << 16);

    asm volatile("tensor_load_to_lds %0, %1, %2, %3"
                 :
                 : "s"(g0), "s"(g1), "s"(g2), "s"(g3)
                 : "memory");
    __builtin_amdgcn_s_wait_tensorcnt(0);
  }
  __syncthreads();

  int wave = tid >> 5;
  int lane = tid & 31;
  int mrow = lane & 15;  // M row / N col owned by this lane
  int khalf = lane >> 4; // which K half this lane holds

  const bf16* arow = &xtile[mrow * DIMC + khalf * 16];  // LDS A rows
  const bf16* w1b = w1t + (size_t)e * HID * DIMC;
  const bf16* w3b = w3t + (size_t)e * HID * DIMC;
  const bf16* w2b = w2t + (size_t)e * DIMC * HID;

  // ---- Phase 1: h/g chunks of 16 columns each, K = DIMC ----
  for (int j = wave; j < HID / 16; j += 8) {
    int h0 = j * 16;
    const bf16* b1row = w1b + (size_t)(h0 + mrow) * DIMC + khalf * 16;
    const bf16* b3row = w3b + (size_t)(h0 + mrow) * DIMC + khalf * 16;
    v8f acch = {};
    v8f accg = {};
#pragma unroll 4
    for (int kb = 0; kb < DIMC / 32; ++kb) {
      int c0 = kb * 32;
      __builtin_prefetch((const void*)(b1row + c0 + 256), 0, 1);
      __builtin_prefetch((const void*)(b3row + c0 + 256), 0, 1);
      v16bf a = load_a16(arow + c0);  // ds_load_b128 x2
      v16bf b1 = *(const v16bf*)(b1row + c0);
      v16bf b3 = *(const v16bf*)(b3row + c0);
      acch = wmma_bf16(a, b1, acch);
      accg = wmma_bf16(a, b3, accg);
    }
#pragma unroll
    for (int r = 0; r < 8; ++r) {
      float hv = acch[r];
      float gv = accg[r];
      float sw = (hv / (1.f + __expf(-hv))) * gv;  // silu(h) * g
      act[(r + khalf * 8) * HID + h0 + mrow] = (bf16)sw;
    }
  }
  __syncthreads();

  // ---- Phase 2: y chunks of 16 output columns, K = HID (A from LDS) ----
  for (int jc = wave; jc < DIMC / 16; jc += 8) {
    int c0 = jc * 16;
    const bf16* brow = w2b + (size_t)(c0 + mrow) * HID + khalf * 16;
    const bf16* aldr = &act[mrow * HID + khalf * 16];
    v8f acc = {};
#pragma unroll 4
    for (int kb = 0; kb < HID / 32; ++kb) {
      int k0 = kb * 32;
      __builtin_prefetch((const void*)(brow + k0 + 256), 0, 1);
      v16bf a = load_a16(aldr + k0);
      v16bf b = *(const v16bf*)(brow + k0);
      acc = wmma_bf16(a, b, acc);
    }
#pragma unroll
    for (int r = 0; r < 8; ++r) {
      int m = r + khalf * 8;
      float val = acc[r] * tws[m];
      atomicAdd(&out[(size_t)toks[m] * DIMC + c0 + mrow], val);
    }
  }
}

// ---------------------------------------------------------------------------
// Launch
// ---------------------------------------------------------------------------
extern "C" void kernel_launch(void* const* d_in, const int* in_sizes, int n_in,
                              void* d_out, int out_size, void* d_ws,
                              size_t ws_size, hipStream_t stream) {
  (void)in_sizes; (void)n_in; (void)ws_size;
  const float* x  = (const float*)d_in[0];
  const float* Wr = (const float*)d_in[1];
  const float* W1 = (const float*)d_in[2];
  const float* W2 = (const float*)d_in[3];
  const float* W3 = (const float*)d_in[4];
  float* out = (float*)d_out;

  char* ws = (char*)d_ws;
  size_t off = 0;
  auto take = [&](size_t bytes) {
    char* p = ws + off;
    off = (off + bytes + 255) & ~(size_t)255;
    return p;
  };
  bf16* xb   = (bf16*)take((size_t)NTOK * DIMC * sizeof(bf16));
  bf16* w1t  = (bf16*)take((size_t)NEXP * HID * DIMC * sizeof(bf16));
  bf16* w3t  = (bf16*)take((size_t)NEXP * HID * DIMC * sizeof(bf16));
  bf16* w2t  = (bf16*)take((size_t)NEXP * DIMC * HID * sizeof(bf16));
  int*   tki = (int*)take((size_t)NTOK * TOPK * sizeof(int));
  float* tkw = (float*)take((size_t)NTOK * TOPK * sizeof(float));
  int*   cnt = (int*)take(NEXP * sizeof(int));
  int*  list = (int*)take((size_t)NEXP * NTOK * sizeof(int));
  float* lwv = (float*)take((size_t)NEXP * NTOK * sizeof(float));

  hipMemsetAsync(d_out, 0, (size_t)out_size * sizeof(float), stream);
  hipMemsetAsync(cnt, 0, NEXP * sizeof(int), stream);

  f32_to_bf16_kernel<<<1024, 256, 0, stream>>>(x, xb, (size_t)NTOK * DIMC);

  dim3 tb(32, 8);
  transpose_bf16_kernel<<<dim3(HID / 32, DIMC / 32, NEXP), tb, 0, stream>>>(
      W1, w1t, DIMC, HID);
  transpose_bf16_kernel<<<dim3(HID / 32, DIMC / 32, NEXP), tb, 0, stream>>>(
      W3, w3t, DIMC, HID);
  transpose_bf16_kernel<<<dim3(DIMC / 32, HID / 32, NEXP), tb, 0, stream>>>(
      W2, w2t, HID, DIMC);

  router_kernel<<<NTOK / 8, 256, 0, stream>>>(x, Wr, tki, tkw);
  scatter_kernel<<<(NTOK * TOPK + 255) / 256, 256, 0, stream>>>(tki, tkw, cnt,
                                                                list, lwv);
  moe_gemm_kernel<<<dim3(NTOK / 16, NEXP), 256, 0, stream>>>(
      xb, w1t, w3t, w2t, cnt, list, lwv, out);
}
